// W4A4O16Linear_59219009077632
// MI455X (gfx1250) — 8-bit matrix operand order repaired and checked by reading, compiled, not run on hardware
//
#include <hip/hip_runtime.h>

typedef _Float16 f16;
typedef __attribute__((ext_vector_type(8))) int   v8i;
typedef __attribute__((ext_vector_type(4))) int   v4i;
typedef __attribute__((ext_vector_type(2))) int   v2i;
typedef __attribute__((ext_vector_type(8))) _Float16 h8;
typedef __attribute__((ext_vector_type(8))) signed char c8;

#define QMINF (-8.0f)
#define QMAXF (7.0f)

__device__ __forceinline__ unsigned enc_f32(float f) {
    int i = __float_as_int(f);
    unsigned u = (unsigned)i;
    return (i < 0) ? ~u : (u | 0x80000000u);
}
__device__ __forceinline__ float dec_f32(unsigned u) {
    int i = (u & 0x80000000u) ? (int)(u & 0x7fffffffu) : (int)~u;
    return __int_as_float(i);
}

__global__ void init_params_kernel(unsigned* pu) {
    if (threadIdx.x == 0) {
        pu[0] = 0xFFFFFFFFu; pu[1] = 0u;
        pu[2] = 0xFFFFFFFFu; pu[3] = 0u;
    }
}

__global__ void minmax_kernel(const f16* __restrict__ x, size_t n8,
                              unsigned* __restrict__ mn, unsigned* __restrict__ mx) {
    float lmin = 3.4e38f, lmax = -3.4e38f;
    size_t i = (size_t)blockIdx.x * blockDim.x + threadIdx.x;
    size_t stride = (size_t)gridDim.x * blockDim.x;
    for (; i < n8; i += stride) {
        h8 v = ((const h8*)x)[i];
        #pragma unroll
        for (int j = 0; j < 8; ++j) {
            float f = (float)v[j];
            lmin = fminf(lmin, f);
            lmax = fmaxf(lmax, f);
        }
    }
    __shared__ float smin[256], smax[256];
    smin[threadIdx.x] = lmin; smax[threadIdx.x] = lmax;
    __syncthreads();
    for (int s = 128; s > 0; s >>= 1) {
        if ((int)threadIdx.x < s) {
            smin[threadIdx.x] = fminf(smin[threadIdx.x], smin[threadIdx.x + s]);
            smax[threadIdx.x] = fmaxf(smax[threadIdx.x], smax[threadIdx.x + s]);
        }
        __syncthreads();
    }
    if (threadIdx.x == 0) {
        atomicMin(mn, enc_f32(smin[0]));
        atomicMax(mx, enc_f32(smax[0]));
    }
}

__global__ void qparams_kernel(unsigned* pu) {
    if (threadIdx.x == 0) {
        float* pf = (float*)pu;
        float xmin = dec_f32(pu[0]), xmax = dec_f32(pu[1]);
        float wmin = dec_f32(pu[2]), wmax = dec_f32(pu[3]);
        float sx = (xmax - xmin) / 15.0f;
        float sw = (wmax - wmin) / 15.0f;
        pf[4] = sx * sw;
        pf[5] = rintf(QMINF - xmin / sx);
        pf[6] = rintf(QMINF - wmin / sw);
        pf[7] = 1.0f / sx;
        pf[8] = 1.0f / sw;
    }
}

__global__ void pack_kernel(const f16* __restrict__ x, signed char* __restrict__ q,
                            size_t n8, const float* __restrict__ p, int sel) {
    const float inv = p[7 + sel];
    const float zp  = p[5 + sel];
    size_t i = (size_t)blockIdx.x * blockDim.x + threadIdx.x;
    size_t stride = (size_t)gridDim.x * blockDim.x;
    for (; i < n8; i += stride) {
        h8 v = ((const h8*)x)[i];
        c8 o;
        #pragma unroll
        for (int j = 0; j < 8; ++j) {
            float r = rintf((float)v[j] * inv) + zp;
            r = fminf(fmaxf(r, QMINF), QMAXF) - zp;
            o[j] = (signed char)(int)r;
        }
        ((c8*)q)[i] = o;
    }
}

#define BM 128
#define BN 128
#define BK 64
#define LDP 80

__device__ __forceinline__ void async_copy16(const signed char* gptr, void* lptr) {
    unsigned long long ga = (unsigned long long)(size_t)gptr;
    unsigned lofs = (unsigned)(size_t)lptr;
    asm volatile("global_load_async_to_lds_b128 %0, %1, off"
                 :: "v"(lofs), "v"(ga) : "memory");
}
__device__ __forceinline__ void wait_async0() {
    asm volatile("s_wait_asynccnt 0x0" ::: "memory");
}

__global__ __launch_bounds__(256) void gemm_kernel(
    const signed char* __restrict__ A,
    const signed char* __restrict__ B,
    const f16* __restrict__ bias,
    f16* __restrict__ out,
    int M, int N, int K, const float* __restrict__ p)
{
    __shared__ signed char As[2][BM][LDP];
    __shared__ signed char Bs[2][BN][LDP];

    const int tid  = threadIdx.x;
    const int lane = tid & 31;
    const int wave = tid >> 5;
    const int wm   = wave & 3;
    const int wn   = wave >> 2;
    const int l15  = lane & 15;
    const int lh   = lane >> 4;

    const int bm0 = blockIdx.y * BM;
    const int bn0 = blockIdx.x * BN;

    const signed char* aptr = A + (size_t)bm0 * K;
    const signed char* bptr = B + (size_t)bn0 * K;

    int srow[2], scol[2];
    #pragma unroll
    for (int it = 0; it < 2; ++it) {
        int v = tid + it * 256;
        srow[it] = v >> 2;
        scol[it] = (v & 3) * 16;
    }

    auto async_stage = [&](int k0, int buf) {
        #pragma unroll
        for (int it = 0; it < 2; ++it) {
            async_copy16(aptr + (size_t)srow[it] * K + k0 + scol[it],
                         &As[buf][srow[it]][scol[it]]);
            async_copy16(bptr + (size_t)srow[it] * K + k0 + scol[it],
                         &Bs[buf][srow[it]][scol[it]]);
        }
    };

    v8i acc[2][4] = {};

    const int mrb = wm * 32;
    const int ncb = wn * 64;

    const int steps = K / BK;
    async_stage(0, 0);

    for (int s = 0; s < steps; ++s) {
        wait_async0();
        __syncthreads();
        if (s + 1 < steps) async_stage((s + 1) * BK, (s + 1) & 1);

        const int buf = s & 1;

        v8i afrag[2];
        #pragma unroll
        for (int mt = 0; mt < 2; ++mt) {
            const signed char* base = &As[buf][mrb + mt * 16 + l15][lh * 16];
            union { v8i v; v2i d[4]; } u;
            u.d[0] = *(const v2i*)(base);
            u.d[1] = *(const v2i*)(base + 8);
            u.d[2] = *(const v2i*)(base + 32);
            u.d[3] = *(const v2i*)(base + 40);
            afrag[mt] = u.v;
        }

        #pragma unroll
        for (int nt = 0; nt < 4; ++nt) {
            const signed char* base = &Bs[buf][ncb + nt * 16 + l15][lh * 16];
            union { v8i v; v4i q[2]; } u;
            u.q[0] = *(const v4i*)(base);
            u.q[1] = *(const v4i*)(base + 32);
            const v8i bfrag = u.v;
            #pragma unroll
            for (int mt = 0; mt < 2; ++mt) {
                acc[mt][nt] = __builtin_amdgcn_wmma_i32_16x16x64_iu8(
                     true, afrag[mt],
                     true, bfrag,
                    acc[mt][nt],  false,  false);
            }
        }
    }

    const float sxw = p[4];
    #pragma unroll
    for (int nt = 0; nt < 4; ++nt) {
        const int gn = bn0 + ncb + nt * 16 + l15;
        const f16 bv = bias[gn];
        #pragma unroll
        for (int mt = 0; mt < 2; ++mt) {
            #pragma unroll
            for (int i = 0; i < 8; ++i) {
                const int gm = bm0 + mrb + mt * 16 + lh * 8 + i;
                f16 r = (f16)(sxw * (float)acc[mt][nt][i]);
                out[(size_t)gm * N + gn] = (f16)(r + bv);
            }
        }
    }
}

extern "C" void kernel_launch(void* const* d_in, const int* in_sizes, int n_in,
                              void* d_out, int out_size, void* d_ws, size_t ws_size,
                              hipStream_t stream) {
    const f16* x    = (const f16*)d_in[0];
    const f16* w    = (const f16*)d_in[1];
    const f16* bias = (const f16*)d_in[2];
    f16* out = (f16*)d_out;

    const size_t nx = (size_t)in_sizes[0];
    const size_t nw = (size_t)in_sizes[1];
    const int N = in_sizes[2];
    const int K = (int)(nw / (size_t)N);
    const int M = (int)(nx / (size_t)K);

    unsigned* pu = (unsigned*)d_ws;
    float*    pf = (float*)d_ws;
    signed char* aq = (signed char*)d_ws + 256;
    signed char* wq = aq + nx;

    init_params_kernel<<<1, 64, 0, stream>>>(pu);
    minmax_kernel<<<2048, 256, 0, stream>>>(x, nx / 8, pu + 0, pu + 1);
    minmax_kernel<<<2048, 256, 0, stream>>>(w, nw / 8, pu + 2, pu + 3);
    qparams_kernel<<<1, 64, 0, stream>>>(pu);
    pack_kernel<<<4096, 256, 0, stream>>>(x, aq, nx / 8, pf, 0);
    pack_kernel<<<4096, 256, 0, stream>>>(w, wq, nw / 8, pf, 1);

    dim3 grid(N / BN, M / BM);
    gemm_kernel<<<grid, 256, 0, stream>>>(aq, wq, bias, out, M, N, K, pf);
}
